// WindowAttention_55937654063207
// MI455X (gfx1250) — compile-verified, matrix operand order checked by reading, not run on hardware
//
#include <hip/hip_runtime.h>
#include <hip/hip_bf16.h>

// ---------------- problem constants ----------------
#define WINS 8192
#define NTOK 49
#define NPAD 64
#define CH   128
#define NH   4
#define HD   32

typedef __attribute__((ext_vector_type(16))) __bf16 v16bf;
typedef __attribute__((ext_vector_type(2)))  __bf16 v2bf;
typedef __attribute__((ext_vector_type(8)))  float  v8f;

// ---------------- LDS layout (padded strides, elements) ----------------
#define XS 136   // x / O row stride (128 + 8)
#define QS 40    // q,k per-head row stride (32 + 8)
#define VS 72    // v^T row stride (64 + 8)
#define SS 72    // scores f32 row stride
#define PS 72    // P bf16 row stride (aliases S)

#define XB_OFF 0
#define QB_OFF (XB_OFF + NPAD * XS * 2)            // 17408
#define KB_OFF (QB_OFF + NH * NPAD * QS * 2)       // 37888
#define VB_OFF (KB_OFF + NH * NPAD * QS * 2)       // 58368
#define SB_OFF (VB_OFF + NH * HD * VS * 2)         // 76800
#define SMEM_BYTES (SB_OFF + NH * NPAD * SS * 4)   // 150528 (< 320KB WGP LDS)

// ---------------- workspace layout (bytes) ----------------
#define WS_QKVWT 0                                  // 384x128 bf16 (transposed)
#define WS_PROJWT (WS_QKVWT + 384 * 128 * 2)        // 128x128 bf16 (transposed)
#define WS_BIASG  (WS_PROJWT + 128 * 128 * 2)       // [H][49][49] f32 gathered bias

// ---------------- helpers ----------------
__device__ __forceinline__ unsigned short f2bf_bits(float f) {
  unsigned u = __float_as_uint(f);
  u += 0x7fffu + ((u >> 16) & 1u);   // round-to-nearest-even
  return (unsigned short)(u >> 16);
}
__device__ __forceinline__ __bf16 f2bf(float f) {
  unsigned short h = f2bf_bits(f);
  __bf16 r;
  __builtin_memcpy(&r, &h, 2);
  return r;
}

// 16-bit A/B fragment loader per CDNA5 ISA 7.12.2:
// lane m = lane&15 selects row; half-wave selects K-halves; VGPR j holds K pair
// k = (j&3)*2 + (j>>2)*16 + (lane>>4)*8. B fragments are loaded from B^T stored
// row-major, which has the identical layout.
__device__ __forceinline__ v16bf load_frag(const __bf16* base, int stride, int lane) {
  const int m = lane & 15;
  const int half = lane >> 4;
  const __bf16* p = base + m * stride + half * 8;
  v16bf f;
#pragma unroll
  for (int j = 0; j < 8; ++j) {
    const int k0 = (j & 3) * 2 + (j >> 2) * 16;
    unsigned u = *reinterpret_cast<const unsigned*>(p + k0);   // 4B-aligned (even strides)
    v2bf t = __builtin_bit_cast(v2bf, u);
    f[2 * j]     = t.x;
    f[2 * j + 1] = t.y;
  }
  return f;
}

__device__ __forceinline__ v8f wmma_bf16(v16bf a, v16bf b, v8f c) {
  // (neg_a, A, neg_b, B, c_mod, C, reuse_a, reuse_b)
  return __builtin_amdgcn_wmma_f32_16x16x32_bf16(false, a, false, b, (short)0, c, false, false);
}

// ---------------- prepass: bf16 weight transpose + bias gather ----------------
__global__ void wa_prepass(const float* __restrict__ qkv_w,     // [128][384]
                           const float* __restrict__ proj_w,    // [128][128]
                           const float* __restrict__ bias_table,// [169][4]
                           const int*   __restrict__ rel_index, // [49*49]
                           __bf16* __restrict__ qkv_wT,         // [384][128]
                           __bf16* __restrict__ proj_wT,        // [128][128]
                           float*  __restrict__ biasg) {        // [4][49][49]
  int idx = blockIdx.x * 256 + threadIdx.x;
  if (idx < 384 * 128) {
    int o = idx >> 7, c = idx & 127;
    qkv_wT[idx] = f2bf(qkv_w[c * 384 + o]);
  } else if (idx < 384 * 128 + 128 * 128) {
    int t = idx - 384 * 128;
    int o = t >> 7, c = t & 127;
    proj_wT[t] = f2bf(proj_w[c * 128 + o]);
  } else if (idx < 384 * 128 + 128 * 128 + NH * NTOK * NTOK) {
    int t = idx - (384 * 128 + 128 * 128);
    int h = t / (NTOK * NTOK);
    int ij = t % (NTOK * NTOK);
    biasg[t] = bias_table[rel_index[ij] * NH + h];
  }
}

// ---------------- main: one window per 8-wave block ----------------
__global__ __launch_bounds__(256, 1) void wa_main(
    const float* __restrict__ x,       // [B][49][128]
    const float* __restrict__ mask,    // [B][1][49][49]
    const float* __restrict__ qkv_b,   // [384]
    const float* __restrict__ proj_b,  // [128]
    const __bf16* __restrict__ qkv_wT, // [384][128]
    const __bf16* __restrict__ proj_wT,// [128][128]
    const float* __restrict__ biasg,   // [4][49][49]
    float* __restrict__ out) {         // [B][49][128]
  extern __shared__ char smem[];
  __bf16* xb   = reinterpret_cast<__bf16*>(smem + XB_OFF);  // x, later O (bf16)
  __bf16* qb   = reinterpret_cast<__bf16*>(smem + QB_OFF);  // [h][64][QS]
  __bf16* kb   = reinterpret_cast<__bf16*>(smem + KB_OFF);  // [h][64][QS]
  __bf16* vb   = reinterpret_cast<__bf16*>(smem + VB_OFF);  // [h][32][VS] (V^T)
  float*  sb   = reinterpret_cast<float*>(smem + SB_OFF);   // [h][64][SS] f32
  __bf16* pbuf = reinterpret_cast<__bf16*>(smem + SB_OFF);  // alias: [h][64][PS] bf16

  const int tid  = threadIdx.x;
  const int lane = tid & 31;
  const int wv   = tid >> 5;
  const int bid  = blockIdx.x;
  const int half = lane >> 4;
  const int ncol = lane & 15;

  // ---- phase 0: stage x -> LDS bf16 (zero-pad rows 49..63) ----
  {
    const float4* xsrc = reinterpret_cast<const float4*>(x + (size_t)bid * NTOK * CH);
#pragma unroll
    for (int i = tid; i < NPAD * (CH / 4); i += 256) {
      int r = i >> 5;
      int c4 = (i & 31) * 4;
      float4 v;
      if (r < NTOK) v = xsrc[r * (CH / 4) + (i & 31)];
      else          v = make_float4(0.f, 0.f, 0.f, 0.f);
      unsigned p0 = ((unsigned)f2bf_bits(v.y) << 16) | f2bf_bits(v.x);
      unsigned p1 = ((unsigned)f2bf_bits(v.w) << 16) | f2bf_bits(v.z);
      unsigned* dst = reinterpret_cast<unsigned*>(xb + r * XS + c4);
      dst[0] = p0; dst[1] = p1;
    }
  }
  __syncthreads();

  // ---- phase 1: QKV = x @ qkv_w^T(T) + qkv_b; scatter into q/k/v^T LDS ----
  const float scale = 0.17677669529663687f;  // HD^-0.5
#pragma unroll
  for (int i = 0; i < 3; ++i) {
    const int nt = wv + i * 8;                 // 24 N-tiles over 8 waves
    // batch all 4 B fragments (global/L2): issue early, maximum latency hiding
    v16bf bfr[4];
#pragma unroll
    for (int ks = 0; ks < 4; ++ks)
      bfr[ks] = load_frag(qkv_wT + nt * 16 * CH + ks * 32, CH, lane);
    v8f acc[4] = {};
#pragma unroll
    for (int ks = 0; ks < 4; ++ks) {           // K = 128 in steps of 32
      // batch the 4 A fragments, then issue 4 back-to-back WMMAs
      v16bf af[4];
#pragma unroll
      for (int mt = 0; mt < 4; ++mt)
        af[mt] = load_frag(xb + mt * 16 * XS + ks * 32, XS, lane);
#pragma unroll
      for (int mt = 0; mt < 4; ++mt)
        acc[mt] = wmma_bf16(af[mt], bfr[ks], acc[mt]);
    }
    const int col = nt * 16 + ncol;            // 0..383
    const int which = col >> 7;                // 0:q 1:k 2:v
    const int rem = col & 127;
    const int h = rem >> 5;
    const int d = rem & 31;
    const float cb = qkv_b[col];
#pragma unroll
    for (int mt = 0; mt < 4; ++mt) {
#pragma unroll
      for (int r = 0; r < 8; ++r) {
        const int m = mt * 16 + r + half * 8;  // token row
        float val = acc[mt][r] + cb;
        if (which == 0)      qb[h * (NPAD * QS) + m * QS + d] = f2bf(val * scale);
        else if (which == 1) kb[h * (NPAD * QS) + m * QS + d] = f2bf(val);
        else                 vb[h * (HD * VS) + d * VS + m]   = f2bf(val);  // transposed
      }
    }
  }
  __syncthreads();

  // ---- phase 2: S = q @ k^T per head (K=32 -> single WMMA per tile) ----
#pragma unroll
  for (int i = 0; i < 2; ++i) {
    const int slot = wv + i * 8;               // 16 (h, nt) slots
    const int h = slot >> 2;
    const int nt = slot & 3;
    v16bf bf = load_frag(kb + h * (NPAD * QS) + nt * 16 * QS, QS, lane);
    v16bf af[4];
#pragma unroll
    for (int mt = 0; mt < 4; ++mt)
      af[mt] = load_frag(qb + h * (NPAD * QS) + mt * 16 * QS, QS, lane);
    v8f c[4] = {};
#pragma unroll
    for (int mt = 0; mt < 4; ++mt)
      c[mt] = wmma_bf16(af[mt], bf, c[mt]);
#pragma unroll
    for (int mt = 0; mt < 4; ++mt)
#pragma unroll
      for (int r = 0; r < 8; ++r) {
        const int m = mt * 16 + r + half * 8;
        sb[h * (NPAD * SS) + m * SS + nt * 16 + ncol] = c[mt][r];
      }
  }
  __syncthreads();

  // ---- phase 3: softmax(S + bias + mask) -> P bf16 (aliases S, barrier-split) ----
  float srow[NTOK];
  float inv = 0.f;
  const bool act = tid < NH * NTOK;            // 196 rows, one per thread
  const int sh = tid / NTOK;
  const int sr = tid - sh * NTOK;
  if (act) {
    const float* sp = sb + sh * (NPAD * SS) + sr * SS;
    const float* bp = biasg + sh * (NTOK * NTOK) + sr * NTOK;
    const float* mp = mask + (size_t)bid * (NTOK * NTOK) + sr * NTOK;
    float mx = -3.0e38f;
#pragma unroll
    for (int j = 0; j < NTOK; ++j) {
      float s = sp[j] + bp[j] + mp[j];
      srow[j] = s;
      mx = fmaxf(mx, s);
    }
    float sum = 0.f;
#pragma unroll
    for (int j = 0; j < NTOK; ++j) {
      float e = __expf(srow[j] - mx);
      srow[j] = e;
      sum += e;
    }
    inv = 1.f / sum;
  }
  __syncthreads();                              // all S reads done before alias writes
  if (act) {
    __bf16* pp = pbuf + sh * (NPAD * PS) + sr * PS;
#pragma unroll
    for (int j = 0; j < NTOK; ++j) pp[j] = f2bf(srow[j] * inv);
#pragma unroll
    for (int j = NTOK; j < NPAD; ++j) pp[j] = f2bf(0.f);   // zero K-padding
  }
  __syncthreads();

  // ---- phase 4: O = P @ V (B fragment from V^T), O reuses xb ----
  {
    const int h = wv >> 1;
    const int nt = wv & 1;
    v16bf bfr[2];
#pragma unroll
    for (int ks = 0; ks < 2; ++ks)
      bfr[ks] = load_frag(vb + h * (HD * VS) + nt * 16 * VS + ks * 32, VS, lane);
    v8f acc[4] = {};
#pragma unroll
    for (int ks = 0; ks < 2; ++ks) {            // K = 64 tokens
      v16bf af[4];
#pragma unroll
      for (int mt = 0; mt < 4; ++mt)
        af[mt] = load_frag(pbuf + h * (NPAD * PS) + mt * 16 * PS + ks * 32, PS, lane);
#pragma unroll
      for (int mt = 0; mt < 4; ++mt)
        acc[mt] = wmma_bf16(af[mt], bfr[ks], acc[mt]);
    }
    const int col = h * HD + nt * 16 + ncol;    // 0..127
#pragma unroll
    for (int mt = 0; mt < 4; ++mt)
#pragma unroll
      for (int r = 0; r < 8; ++r) {
        const int m = mt * 16 + r + half * 8;
        xb[m * XS + col] = f2bf(acc[mt][r]);
      }
  }
  __syncthreads();

  // ---- phase 5: out = O @ proj_w^T(T) + proj_b ----
  {
    const int nt = wv;                          // 8 N-tiles, one per wave
    v16bf bfr[4];
#pragma unroll
    for (int ks = 0; ks < 4; ++ks)
      bfr[ks] = load_frag(proj_wT + nt * 16 * CH + ks * 32, CH, lane);
    v8f acc[4] = {};
#pragma unroll
    for (int ks = 0; ks < 4; ++ks) {
      v16bf af[4];
#pragma unroll
      for (int mt = 0; mt < 4; ++mt)
        af[mt] = load_frag(xb + mt * 16 * XS + ks * 32, XS, lane);
#pragma unroll
      for (int mt = 0; mt < 4; ++mt)
        acc[mt] = wmma_bf16(af[mt], bfr[ks], acc[mt]);
    }
    const int colo = nt * 16 + ncol;
    const float cb = proj_b[colo];
    float* op = out + (size_t)bid * (NTOK * CH);
#pragma unroll
    for (int mt = 0; mt < 4; ++mt)
#pragma unroll
      for (int r = 0; r < 8; ++r) {
        const int m = mt * 16 + r + half * 8;
        if (m < NTOK) op[m * CH + colo] = acc[mt][r] + cb;
      }
  }
}

extern "C" void kernel_launch(void* const* d_in, const int* in_sizes, int n_in,
                              void* d_out, int out_size, void* d_ws, size_t ws_size,
                              hipStream_t stream) {
  const float* x          = (const float*)d_in[0];
  const float* mask       = (const float*)d_in[1];
  const float* qkv_w      = (const float*)d_in[2];
  const float* qkv_b      = (const float*)d_in[3];
  const float* bias_table = (const float*)d_in[4];
  const float* proj_w     = (const float*)d_in[5];
  const float* proj_b     = (const float*)d_in[6];
  const int*   rel_index  = (const int*)d_in[7];

  char* ws = (char*)d_ws;
  __bf16* qkv_wT  = (__bf16*)(ws + WS_QKVWT);
  __bf16* proj_wT = (__bf16*)(ws + WS_PROJWT);
  float*  biasg   = (float*)(ws + WS_BIASG);
  float*  out     = (float*)d_out;

  const int total = 384 * 128 + 128 * 128 + NH * NTOK * NTOK;
  wa_prepass<<<(total + 255) / 256, 256, 0, stream>>>(
      qkv_w, proj_w, bias_table, rel_index, qkv_wT, proj_wT, biasg);
  wa_main<<<WINS, 256, SMEM_BYTES, stream>>>(
      x, mask, qkv_b, proj_b, qkv_wT, proj_wT, biasg, out);
}
